// MultiHeadedAttention_90933047591300
// MI455X (gfx1250) — compile-verified, matrix operand order checked by reading, not run on hardware
//
#include <hip/hip_runtime.h>
#include <hip/hip_bf16.h>

typedef __attribute__((ext_vector_type(8)))  _Float16 v8h;
typedef __attribute__((ext_vector_type(16))) _Float16 v16h;
typedef __attribute__((ext_vector_type(8)))  float    v8f;
typedef __attribute__((ext_vector_type(4)))  unsigned int u32x4;
typedef __attribute__((ext_vector_type(4)))  int      i32x4;
typedef __attribute__((ext_vector_type(8)))  int      i32x8;

#define NUM_HEADS 16
#define HEAD_DIM  64
#define EMBED     1024
#define BATCH     2
#define SEQ       2048
#define QKV_COLS  (3 * NUM_HEADS * HEAD_DIM)   // 3072
#define ROWS      (BATCH * SEQ)                // 4096

#if __has_builtin(__builtin_amdgcn_tensor_load_to_lds)
#define HAVE_TDM 1
#else
#define HAVE_TDM 0
#endif

// ---------------------------------------------------------------------------
// Address-space helpers (CK-style explicit AS casts, accepted in HIP mode)
// ---------------------------------------------------------------------------
__device__ __forceinline__ unsigned lds_off(const void* p) {
  return (unsigned)(unsigned long long)
      (__attribute__((address_space(3))) const void*)p;
}

__device__ __forceinline__ void wait_tensorcnt0() {
#if __has_builtin(__builtin_amdgcn_s_wait_tensorcnt)
  __builtin_amdgcn_s_wait_tensorcnt(0);
#else
  asm volatile("s_wait_tensorcnt 0x0" ::: "memory");
#endif
}

// One async 16-byte lane copy: global -> LDS (tracked by ASYNCcnt).
__device__ __forceinline__ void async_copy_b128(unsigned ldsAddr,
                                                const void* gsrc) {
  asm volatile("global_load_async_to_lds_b128 %0, %1, off"
               :: "v"(ldsAddr), "v"((unsigned long long)gsrc)
               : "memory");
}
__device__ __forceinline__ void wait_asynccnt0() {
  asm volatile("s_wait_asynccnt 0x0" ::: "memory");
}

#if HAVE_TDM
// TDM 2-D f16 tile load: tile_w elements/row (must be 64 here: padding config
// assumes 128B rows -> pad_interval=32 DWORDs, pad_amount=4 DWORDs => LDS row
// pitch 144B = 72 halfs).  D# layout per CDNA5 ISA 8.3/8.4.
// This toolchain exposes the 6-arg builtin:
//   (u32x4 g0, i32x8 g1, i32x4 g2, i32x4 g3, i32x8 pad, i32 cpol)
__device__ __forceinline__ void tdm_load_2d_f16(
    unsigned ldsAddr, const _Float16* gsrc, unsigned tile_w, unsigned tile_h,
    unsigned tensor_w, unsigned tensor_h, unsigned row_stride_elems) {
  unsigned long long ga = (unsigned long long)gsrc;
  u32x4 g0;
  g0[0] = 1u;                                           // count=1, user mode
  g0[1] = ldsAddr;                                      // lds_addr (bytes)
  g0[2] = (unsigned)(ga & 0xffffffffull);               // global_addr[31:0]
  g0[3] = (unsigned)((ga >> 32) & 0x1ffffffull)         // global_addr[56:32]
          | (2u << 30);                                 // type=2 ("image")
  i32x8 g1;
  g1[0] = (int)((1u << 16)      // data_size = 2 bytes
              | (1u << 20)      // pad_enable
              | (4u << 22)      // pad_interval: 32 DWORDs (=128B tile row)
              | (3u << 25));    // pad_amount: 4 DWORDs (=16B)
  g1[1] = (int)((tensor_w & 0xffffu) << 16);            // tensor_dim0[15:0]
  g1[2] = (int)(((tensor_w >> 16) & 0xffffu)            // tensor_dim0[31:16]
              | ((tensor_h & 0xffffu) << 16));          // tensor_dim1[15:0]
  g1[3] = (int)(((tensor_h >> 16) & 0xffffu)            // tensor_dim1[31:16]
              | ((tile_w & 0xffffu) << 16));            // tile_dim0
  g1[4] = (int)(tile_h & 0xffffu);                      // tile_dim1 (dim2=0)
  g1[5] = (int)row_stride_elems;                        // dim0_stride[31:0]
  g1[6] = 0;                                            // stride hi / dim1_stride
  g1[7] = 0;
  i32x4 z4 = {0, 0, 0, 0};
  i32x8 z8 = {0, 0, 0, 0, 0, 0, 0, 0};
  __builtin_amdgcn_tensor_load_to_lds(g0, g1, z4, z4, z8, 0);
}
#endif

// ---------------------------------------------------------------------------
// WMMA helpers
// ---------------------------------------------------------------------------
__device__ __forceinline__ v8f wmma16x16x32(v16h a, v16h b, v8f c) {
  return __builtin_amdgcn_wmma_f32_16x16x32_f16(
      false, a, false, b, (short)0, c, false, false);
}

// 16x32 fragment from a padded row-major LDS tile.  CDNA5 16-bit A layout:
// lanes 0-15: K={0..7,16..23}; lanes 16-31: K={8..15,24..31} -> 2x b128.
__device__ __forceinline__ v16h frag16(const _Float16* rowp, int kofs, int lane) {
  const _Float16* p = rowp + kofs + ((lane < 16) ? 0 : 8);
  v8h lo = *(const v8h*)(p);
  v8h hi = *(const v8h*)(p + 16);
  return __builtin_shufflevector(lo, hi, 0, 1, 2, 3, 4, 5, 6, 7,
                                         8, 9, 10, 11, 12, 13, 14, 15);
}

// ---------------------------------------------------------------------------
// GEMM + bias:  C[M,N] = A[M,K] @ B[K,N] + bias[N]
// AT = float (convert while staging) or _Float16 (async b128 copy to LDS).
// OT = float or _Float16.  Tile 64x64x32, 256 threads = 8 waves (4x2).
// ---------------------------------------------------------------------------
#define GT_M 64
#define GT_N 64
#define GT_K 32
#define LDSA 40   // 80B row pitch: 16B aligned, bank-spread

template <typename AT, typename OT>
__global__ __launch_bounds__(256) void gemm_bias_wmma(
    const AT* __restrict__ A, const float* __restrict__ B,
    const float* __restrict__ bias, OT* __restrict__ C,
    int M, int N, int K) {
  __shared__ _Float16 As[GT_M][LDSA];
  __shared__ _Float16 Bs[GT_N][LDSA];   // transposed: Bs[n][k]

  const int tid  = threadIdx.x;
  const int lane = tid & 31;
  const int wave = tid >> 5;
  const int wr = wave >> 1;
  const int wc = wave & 1;
  const int m0 = blockIdx.y * GT_M;
  const int n0 = blockIdx.x * GT_N;

  v8f acc0 = {};
  v8f acc1 = {};

  for (int k0 = 0; k0 < K; k0 += GT_K) {
    // ---- stage A tile (64x32) ----
    if constexpr (sizeof(AT) == 4) {
#pragma unroll
      for (int i = 0; i < 2; ++i) {
        int idx = tid + i * 256;
        int r   = idx >> 3;
        int c4  = (idx & 7) << 2;
        float4 v = *(const float4*)(A + (size_t)(m0 + r) * K + k0 + c4);
        _Float16* d = &As[r][c4];
        d[0] = (_Float16)v.x; d[1] = (_Float16)v.y;
        d[2] = (_Float16)v.z; d[3] = (_Float16)v.w;
      }
    } else {
      // f16 source: 256 x 16B chunks, one async copy per thread
      int r  = tid >> 2;
      int c8 = (tid & 3) << 3;
      async_copy_b128(lds_off(&As[r][c8]),
                      A + (size_t)(m0 + r) * K + k0 + c8);
    }
    // ---- stage B tile (32x64 fp32) transposed into Bs[n][k] ----
#pragma unroll
    for (int i = 0; i < 2; ++i) {
      int idx = tid + i * 256;
      int r   = idx >> 4;
      int c4  = (idx & 15) << 2;
      float4 v = *(const float4*)(B + (size_t)(k0 + r) * N + n0 + c4);
      Bs[c4 + 0][r] = (_Float16)v.x;
      Bs[c4 + 1][r] = (_Float16)v.y;
      Bs[c4 + 2][r] = (_Float16)v.z;
      Bs[c4 + 3][r] = (_Float16)v.w;
    }
    if constexpr (sizeof(AT) == 2) wait_asynccnt0();
    __syncthreads();

    v16h a  = frag16(&As[wr * 16 + (lane & 15)][0], 0, lane);
    v16h b0 = frag16(&Bs[wc * 32 + (lane & 15)][0], 0, lane);
    v16h b1 = frag16(&Bs[wc * 32 + 16 + (lane & 15)][0], 0, lane);
    acc0 = wmma16x16x32(a, b0, acc0);
    acc1 = wmma16x16x32(a, b1, acc1);
    __syncthreads();
  }

  const int col0  = n0 + wc * 32 + (lane & 15);
  const int rbase = m0 + wr * 16 + ((lane < 16) ? 0 : 8);
  const float bi0 = bias[col0];
  const float bi1 = bias[col0 + 16];
#pragma unroll
  for (int j = 0; j < 8; ++j) {
    C[(size_t)(rbase + j) * N + col0]      = (OT)(acc0[j] + bi0);
    C[(size_t)(rbase + j) * N + col0 + 16] = (OT)(acc1[j] + bi1);
  }
}

// ---------------------------------------------------------------------------
// Flash attention over the f16 qkv buffer [ROWS, 3072].
// Grid: (SEQ/128, BATCH*NUM_HEADS).  8 waves; each wave owns 16 query rows
// and the full 64-wide key block, so softmax stats reduce within 16-lane
// halves.  Q/K tiles loaded by the Tensor Data Mover (wave 0) when available.
// ---------------------------------------------------------------------------
#define QBLK 128
#define KBLK 64
#define LDSQ 72   // 144B row pitch (matches TDM pad config)

__global__ __launch_bounds__(256) void attn_flash_wmma(
    const _Float16* __restrict__ qkv, _Float16* __restrict__ ctx) {
  __shared__ _Float16 Qs[QBLK][LDSQ];      // [q_row][d]
  __shared__ _Float16 Ks[KBLK][LDSQ];      // [key][d]
  __shared__ _Float16 Vs[HEAD_DIM][LDSQ];  // transposed: [d][key]
  __shared__ _Float16 Ps[8][16][LDSQ];     // per-wave P slab [q_row][key]

  const int tid  = threadIdx.x;
  const int lane = tid & 31;
  const int wave = tid >> 5;
  const int b  = blockIdx.y >> 4;
  const int h  = blockIdx.y & 15;
  const int q0 = blockIdx.x * QBLK;

  const size_t rowstride = QKV_COLS;
  const _Float16* gq = qkv + (size_t)(b * SEQ + q0) * rowstride + h * HEAD_DIM;

  // ---- stage Q block (128x64 f16) ----
#if HAVE_TDM
  if (wave == 0)
    tdm_load_2d_f16(lds_off(&Qs[0][0]), gq, HEAD_DIM, QBLK,
                    QKV_COLS, ROWS, QKV_COLS);
#else
#pragma unroll
  for (int i = 0; i < 4; ++i) {
    int idx = tid + i * 256;           // 0..1023 chunks of 8 halfs
    int r   = idx >> 3;
    int c8  = (idx & 7) << 3;
    *(v8h*)&Qs[r][c8] = *(const v8h*)(gq + (size_t)r * rowstride + c8);
  }
#endif

  float mrun[8], lrun[8];
  v8f o[4] = {};
#pragma unroll
  for (int j = 0; j < 8; ++j) { mrun[j] = -1e30f; lrun[j] = 0.f; }

  const int qrow = wave * 16 + (lane & 15);

  for (int kb = 0; kb < SEQ; kb += KBLK) {
    const _Float16* gk = qkv + (size_t)(b * SEQ + kb) * rowstride +
                         NUM_HEADS * HEAD_DIM + h * HEAD_DIM;
    const _Float16* gv = gk + NUM_HEADS * HEAD_DIM;

    // ---- stage K (TDM) ----
#if HAVE_TDM
    if (wave == 0)
      tdm_load_2d_f16(lds_off(&Ks[0][0]), gk, HEAD_DIM, KBLK,
                      QKV_COLS, ROWS, QKV_COLS);
#else
#pragma unroll
    for (int i = 0; i < 2; ++i) {
      int idx = tid + i * 256;
      int r   = idx >> 3;
      int c8  = (idx & 7) << 3;
      *(v8h*)&Ks[r][c8] = *(const v8h*)(gk + (size_t)r * rowstride + c8);
    }
#endif
    // ---- stage V transposed: Vs[d][key] (b128 load + 8x ds_store_b16) ----
#pragma unroll
    for (int i = 0; i < 2; ++i) {
      int idx = tid + i * 256;         // 0..511 chunks
      int r   = idx >> 3;              // key row
      int c8  = (idx & 7) << 3;        // d offset
      v8h v = *(const v8h*)(gv + (size_t)r * rowstride + c8);
#pragma unroll
      for (int j = 0; j < 8; ++j) Vs[c8 + j][r] = v[j];
    }
#if HAVE_TDM
    if (wave == 0) wait_tensorcnt0();  // covers Q (first iter) + K
#endif
    __syncthreads();

    // ---- scores = Q @ K^T / sqrt(d) : 8 WMMAs ----
    v16h qa0 = frag16(&Qs[qrow][0], 0, lane);
    v16h qa1 = frag16(&Qs[qrow][0], 32, lane);
    v8f sc[4];
#pragma unroll
    for (int n = 0; n < 4; ++n) {
      int krow = n * 16 + (lane & 15);
      v16h kb0 = frag16(&Ks[krow][0], 0, lane);
      v16h kb1 = frag16(&Ks[krow][0], 32, lane);
      v8f s = {};
      s = wmma16x16x32(qa0, kb0, s);
      s = wmma16x16x32(qa1, kb1, s);
      sc[n] = s * 0.125f;
    }

    // ---- online softmax (stats per VGPR-row, reduce over 16-lane half) ----
#pragma unroll
    for (int j = 0; j < 8; ++j) {
      float mx = fmaxf(fmaxf(sc[0][j], sc[1][j]), fmaxf(sc[2][j], sc[3][j]));
      mx = fmaxf(mx, __shfl_xor(mx, 1, 32));
      mx = fmaxf(mx, __shfl_xor(mx, 2, 32));
      mx = fmaxf(mx, __shfl_xor(mx, 4, 32));
      mx = fmaxf(mx, __shfl_xor(mx, 8, 32));
      float mnew = fmaxf(mrun[j], mx);
      float al   = __expf(mrun[j] - mnew);
      mrun[j] = mnew;
      float rs = 0.f;
#pragma unroll
      for (int n = 0; n < 4; ++n) {
        float p = __expf(sc[n][j] - mnew);
        sc[n][j] = p;
        rs += p;
      }
      rs += __shfl_xor(rs, 1, 32);
      rs += __shfl_xor(rs, 2, 32);
      rs += __shfl_xor(rs, 4, 32);
      rs += __shfl_xor(rs, 8, 32);
      lrun[j] = lrun[j] * al + rs;
#pragma unroll
      for (int n = 0; n < 4; ++n) o[n][j] *= al;
    }

    // ---- re-layout P via per-wave LDS slab (same-wave DS ops in order) ----
    {
      _Float16* pw = &Ps[wave][0][0];
      const int cc   = lane & 15;
      const int rofs = (lane < 16) ? 0 : 8;
#pragma unroll
      for (int n = 0; n < 4; ++n)
#pragma unroll
        for (int j = 0; j < 8; ++j)
          pw[(size_t)(j + rofs) * LDSQ + n * 16 + cc] = (_Float16)sc[n][j];
    }

    // ---- ctx += P @ V : 8 WMMAs ----
    v16h pa0 = frag16(&Ps[wave][lane & 15][0], 0, lane);
    v16h pa1 = frag16(&Ps[wave][lane & 15][0], 32, lane);
#pragma unroll
    for (int n = 0; n < 4; ++n) {
      int drow = n * 16 + (lane & 15);
      v16h vb0 = frag16(&Vs[drow][0], 0, lane);
      v16h vb1 = frag16(&Vs[drow][0], 32, lane);
      o[n] = wmma16x16x32(pa0, vb0, o[n]);
      o[n] = wmma16x16x32(pa1, vb1, o[n]);
    }
    __syncthreads();
  }

  // ---- epilogue: ctx[b, s, h*d] = (f16)(o / l) ----
  const int rbase = b * SEQ + q0 + wave * 16 + ((lane < 16) ? 0 : 8);
#pragma unroll
  for (int j = 0; j < 8; ++j) {
    float inv = 1.0f / lrun[j];
#pragma unroll
    for (int n = 0; n < 4; ++n) {
      int dcol = h * HEAD_DIM + n * 16 + (lane & 15);
      ctx[(size_t)(rbase + j) * EMBED + dcol] = (_Float16)(o[n][j] * inv);
    }
  }
}

// ---------------------------------------------------------------------------
// Launch
// ---------------------------------------------------------------------------
extern "C" void kernel_launch(void* const* d_in, const int* in_sizes, int n_in,
                              void* d_out, int out_size, void* d_ws, size_t ws_size,
                              hipStream_t stream) {
  const float* x     = (const float*)d_in[0];   // [B, S, E]
  const float* w_qkv = (const float*)d_in[1];   // [E, 3*H*D]
  const float* b_qkv = (const float*)d_in[2];
  const float* w_o   = (const float*)d_in[3];   // [H*D, E]
  const float* b_o   = (const float*)d_in[4];
  float* out = (float*)d_out;                   // [B, S, E]

  _Float16* qkvbuf = (_Float16*)d_ws;                      // ROWS x 3072 f16
  _Float16* ctxbuf = qkvbuf + (size_t)ROWS * QKV_COLS;     // ROWS x 1024 f16

  // 1) qkv = f16(x @ w_qkv + b_qkv)
  dim3 g1(QKV_COLS / GT_N, ROWS / GT_M);
  gemm_bias_wmma<float, _Float16><<<g1, 256, 0, stream>>>(
      x, w_qkv, b_qkv, qkvbuf, ROWS, QKV_COLS, EMBED);

  // 2) flash attention -> ctx f16 [b, s, h*d]
  dim3 g2(SEQ / QBLK, BATCH * NUM_HEADS);
  attn_flash_wmma<<<g2, 256, 0, stream>>>(qkvbuf, ctxbuf);

  // 3) out = ctx @ w_o + b_o   (A staged via async b128 copies)
  dim3 g3(EMBED / GT_N, ROWS / GT_M);
  gemm_bias_wmma<_Float16, float><<<g3, 256, 0, stream>>>(
      ctxbuf, w_o, b_o, out, ROWS, EMBED, EMBED);
}
